// Attention_89824946028754
// MI455X (gfx1250) — matrix operand order repaired and checked by reading, not compiled, not run on hardware
//
#include <hip/hip_runtime.h>

typedef unsigned short u16;
typedef unsigned long long u64;
typedef __attribute__((ext_vector_type(8)))  unsigned short u16x8;
typedef __attribute__((ext_vector_type(16))) __bf16        v16bf;
typedef __attribute__((ext_vector_type(8)))  float          v8f;

#define B_  4
#define S_  1024
#define DIM_ 1024
#define H_  16
#define HD_ 64

// ---------- bf16 helpers (raw u16 storage, RNE convert) ----------
__device__ __forceinline__ u16 f2bf_raw(float f) {
    union { float f; unsigned u; } v; v.f = f;
    unsigned r = v.u + 0x7FFFu + ((v.u >> 16) & 1u);
    return (u16)(r >> 16);
}

union ABFrag { v16bf v; u16x8 h[2]; };

// A fragment: 16x32 bf16, lane c16 = row M; elem j: K = k0 + half*8 + (j&7) + (j>=8 ? 16 : 0)
__device__ __forceinline__ v16bf load_a_frag(const u16* row, int k0, int half) {
    ABFrag f;
    f.h[0] = *(const u16x8*)(row + k0 + half * 8);
    f.h[1] = *(const u16x8*)(row + k0 + 16 + half * 8);
    return f.v;
}
// B fragment: 32x16 bf16, lane c16 = col N; elem j: K = k0 + half*8 + (j&7) + (j>=8 ? 16 : 0)  (row = transposed source)
__device__ __forceinline__ v16bf load_b_frag(const u16* row, int k0, int half) {
    ABFrag f;
    f.h[0] = *(const u16x8*)(row + k0 + half * 8);
    f.h[1] = *(const u16x8*)(row + k0 + 16 + half * 8);
    return f.v;
}

__device__ __forceinline__ v8f wmma_bf16(v16bf a, v16bf b, v8f c) {
    return __builtin_amdgcn_wmma_f32_16x16x32_bf16(false, a, false, b, (short)0, c, false, false);
}

// Async-stage one contiguous 32KB block (16 rows x 1024 bf16) from global into LDS.
// 2048 x 16B chunks spread over 128 threads; tracked by ASYNCcnt.
__device__ __forceinline__ void stage_a_tile_async(const u16* __restrict__ gsrc, u16* lds_dst) {
    for (int i = threadIdx.x; i < 2048; i += 128) {
        unsigned lds_off = (unsigned)reinterpret_cast<u64>(lds_dst + i * 8);
        u64 gaddr = reinterpret_cast<u64>(gsrc + i * 8);
        asm volatile("global_load_async_to_lds_b128 %0, %1, off"
                     :: "v"(lds_off), "v"(gaddr) : "memory");
    }
    asm volatile("s_wait_asynccnt 0x0" ::: "memory");
    __syncthreads();
}

// ---------- conversion kernels ----------
__global__ void cvt_bf16_kernel(const float* __restrict__ in, u16* __restrict__ out, int n) {
    int i = blockIdx.x * blockDim.x + threadIdx.x;
    if (i < n) out[i] = f2bf_raw(in[i]);
}
// 1024x1024 transpose-convert: out[n][k] = bf16(in[k][n])
__global__ void cvt_t_bf16_kernel(const float* __restrict__ in, u16* __restrict__ out) {
    int i = blockIdx.x * blockDim.x + threadIdx.x;
    int n = i >> 10, k = i & 1023;
    out[i] = f2bf_raw(in[k * DIM_ + n]);
}

// ---------- QKV projection + bias + RoPE ----------
// block = 128 (4 waves); grid.x = 256 M-tiles; grid.y = 48: mat = y>>4, ntg = y&15,
// wave w computes N-tile nt = ntg*4 + w. A tile shared via async-staged LDS.
__global__ __launch_bounds__(128) void qkv_rope_kernel(
    const u16* __restrict__ xb,
    const u16* __restrict__ wqt, const u16* __restrict__ wkt, const u16* __restrict__ wvt,
    const float* __restrict__ bq, const float* __restrict__ bk, const float* __restrict__ bv,
    u16* __restrict__ qout,   // [b][h][s][64]
    u16* __restrict__ kout,   // [b][h][s][64]
    u16* __restrict__ vtout)  // [b][h][64][s]  (transposed V)
{
    __shared__ u16 Atile[16 * DIM_];   // 32 KB

    const int lane = threadIdx.x & 31, wave = threadIdx.x >> 5;
    const int half = lane >> 4, c16 = lane & 15;
    const int mat = blockIdx.y >> 4, nt = (blockIdx.y & 15) * 4 + wave;

    stage_a_tile_async(xb + blockIdx.x * 16 * DIM_, Atile);

    const u16*  wt   = (mat == 0) ? wqt : (mat == 1) ? wkt : wvt;
    const float* bias = (mat == 0) ? bq : (mat == 1) ? bk : bv;

    const u16* arow = Atile + c16 * DIM_;              // ds_load_b128 fragments
    const u16* brow = wt + (nt * 16 + c16) * DIM_;

    v8f acc = {};
    for (int k0 = 0; k0 < DIM_; k0 += 32)
        acc = wmma_bf16(load_a_frag(arow, k0, half), load_b_frag(brow, k0, half), acc);

    const int col = nt * 16 + c16;
    const float bcol = bias[col];
    const int h = col >> 6, d = col & 63;

    for (int r = 0; r < 8; ++r) {
        const int m = half * 8 + r;
        const int g = blockIdx.x * 16 + m;             // flat token index
        const int b = g >> 10, s = g & 1023;
        float v = acc[r] + bcol;
        if (mat < 2) {
            // RoPE: pair (2i, 2i+1); partner column = col^1 lives in lane^1 (same row)
            float p = __shfl_xor(v, 1);
            const int   i   = d >> 1;
            const float ang = (float)s * __powf(10000.f, -(float)(2 * i) / 64.f);
            float sn, cs; __sincosf(ang, &sn, &cs);
            v = (d & 1) ? (v * cs + p * sn) : (v * cs - p * sn);
            u16* dst = mat ? kout : qout;
            dst[(((b * H_ + h) * S_) + s) * HD_ + d] = f2bf_raw(v);
        } else {
            vtout[((b * H_ + h) * HD_ + d) * S_ + s] = f2bf_raw(v);
        }
    }
}

// ---------- attention: one wave per (b,h, 16-query tile) ----------
// grid.x = B*H = 64, grid.y = S/16 = 64, block = 32, dyn smem = 16*1024*4 + 16*1024*2
__global__ __launch_bounds__(32) void attn_kernel(
    const u16* __restrict__ qb, const u16* __restrict__ kb, const u16* __restrict__ vt,
    const unsigned char* __restrict__ mask, u16* __restrict__ att)
{
    extern __shared__ char smem[];
    float* sc = (float*)smem;                       // [16][1024] scores
    u16*   pb = (u16*)(smem + 16 * 1024 * 4);       // [16][1024] bf16 probabilities

    const int lane = threadIdx.x, half = lane >> 4, c16 = lane & 15;
    const int bh = blockIdx.x, b = bh >> 4, h = bh & 15, qt = blockIdx.y;

    const u16* qrow = qb + (bh * S_ + qt * 16 + c16) * HD_;
    const v16bf qa0 = load_a_frag(qrow, 0,  half);
    const v16bf qa1 = load_a_frag(qrow, 32, half);
    const float scale = 0.125f;                     // 1/sqrt(64)

    for (int kt = 0; kt < 64; ++kt) {
        const u16* krow = kb + (bh * S_ + kt * 16 + c16) * HD_;   // B[d][n] = K[n][d]
        v8f acc = {};
        acc = wmma_bf16(qa0, load_b_frag(krow, 0,  half), acc);
        acc = wmma_bf16(qa1, load_b_frag(krow, 32, half), acc);
        const int kpos = kt * 16 + c16;
        const bool ok = mask[b * S_ + kpos] != 0;
        for (int r = 0; r < 8; ++r) {
            const int m = half * 8 + r;
            sc[m * 1024 + kpos] = ok ? acc[r] * scale : -1e30f;
        }
    }
    __syncthreads();

    // softmax: 2 lanes per row (lane, lane+16), 512 cols each
    {
        const int row = c16, part = half;
        float mx = -1e30f;
        for (int k = part * 512; k < part * 512 + 512; ++k) mx = fmaxf(mx, sc[row * 1024 + k]);
        mx = fmaxf(mx, __shfl_xor(mx, 16));
        float sum = 0.f;
        for (int k = part * 512; k < part * 512 + 512; ++k) sum += __expf(sc[row * 1024 + k] - mx);
        sum += __shfl_xor(sum, 16);
        const float inv = 1.f / sum;
        for (int k = part * 512; k < part * 512 + 512; ++k)
            pb[row * 1024 + k] = f2bf_raw(__expf(sc[row * 1024 + k] - mx) * inv);
    }
    __syncthreads();

    // out = P (16x1024) x V (1024x64), four 16-col tiles
    const u16* prow = pb + c16 * 1024;
    for (int nt = 0; nt < 4; ++nt) {
        const u16* vrow = vt + (bh * HD_ + nt * 16 + c16) * S_;   // Vt row: contiguous in k
        v8f acc = {};
        for (int k0 = 0; k0 < S_; k0 += 32)
            acc = wmma_bf16(load_a_frag(prow, k0, half), load_b_frag(vrow, k0, half), acc);
        const int d = nt * 16 + c16;
        for (int r = 0; r < 8; ++r) {
            const int s = qt * 16 + half * 8 + r;
            att[(b * S_ + s) * DIM_ + h * HD_ + d] = f2bf_raw(acc[r]);
        }
    }
}

// ---------- output projection + bias + mask ----------
// block = 128 (4 waves); grid.x = 256 M-tiles, grid.y = 16; wave w: nt = y*4 + w
__global__ __launch_bounds__(128) void oproj_kernel(
    const u16* __restrict__ att, const u16* __restrict__ wot,
    const float* __restrict__ bo, const unsigned char* __restrict__ mask,
    float* __restrict__ out)
{
    __shared__ u16 Atile[16 * DIM_];   // 32 KB

    const int lane = threadIdx.x & 31, wave = threadIdx.x >> 5;
    const int half = lane >> 4, c16 = lane & 15;
    const int nt = blockIdx.y * 4 + wave;

    stage_a_tile_async(att + blockIdx.x * 16 * DIM_, Atile);

    const u16* arow = Atile + c16 * DIM_;
    const u16* brow = wot + (nt * 16 + c16) * DIM_;
    v8f acc = {};
    for (int k0 = 0; k0 < DIM_; k0 += 32)
        acc = wmma_bf16(load_a_frag(arow, k0, half), load_b_frag(brow, k0, half), acc);

    const int col = nt * 16 + c16;
    const float bcol = bo[col];
    for (int r = 0; r < 8; ++r) {
        const int g = blockIdx.x * 16 + half * 8 + r;
        const int b = g >> 10, s = g & 1023;
        const float v = acc[r] + bcol;
        out[g * DIM_ + col] = mask[b * S_ + s] ? v : 0.f;
    }
}

extern "C" void kernel_launch(void* const* d_in, const int* in_sizes, int n_in,
                              void* d_out, int out_size, void* d_ws, size_t ws_size,
                              hipStream_t stream) {
    const float* x    = (const float*)d_in[0];
    const unsigned char* mask = (const unsigned char*)d_in[1];
    // d_in[2] = freqs (recomputed on device, unused)
    const float* Wq = (const float*)d_in[3];
    const float* bq = (const float*)d_in[4];
    const float* Wk = (const float*)d_in[5];
    const float* bk = (const float*)d_in[6];
    const float* Wv = (const float*)d_in[7];
    const float* bv = (const float*)d_in[8];
    const float* Wo = (const float*)d_in[9];
    const float* bo = (const float*)d_in[10];
    float* out = (float*)d_out;

    // workspace layout (bytes)
    char* ws = (char*)d_ws;
    const size_t SZ_XB = (size_t)B_ * S_ * DIM_ * 2;   // 8 MB
    const size_t SZ_W  = (size_t)DIM_ * DIM_ * 2;      // 2 MB each
    u16* xb  = (u16*)(ws);
    u16* wqt = (u16*)(ws + SZ_XB);
    u16* wkt = (u16*)(ws + SZ_XB + SZ_W);
    u16* wvt = (u16*)(ws + SZ_XB + 2 * SZ_W);
    u16* wot = (u16*)(ws + SZ_XB + 3 * SZ_W);
    u16* qbf = (u16*)(ws + SZ_XB + 4 * SZ_W);
    u16* kbf = (u16*)(ws + SZ_XB + 4 * SZ_W + SZ_XB);
    u16* vtb = (u16*)(ws + SZ_XB + 4 * SZ_W + 2 * SZ_XB);
    u16* att = (u16*)(ws + SZ_XB + 4 * SZ_W + 3 * SZ_XB);

    const int NX = B_ * S_ * DIM_;                     // 4,194,304
    cvt_bf16_kernel<<<(NX + 255) / 256, 256, 0, stream>>>(x, xb, NX);
    const int NW = DIM_ * DIM_;
    cvt_t_bf16_kernel<<<NW / 256, 256, 0, stream>>>(Wq, wqt);
    cvt_t_bf16_kernel<<<NW / 256, 256, 0, stream>>>(Wk, wkt);
    cvt_t_bf16_kernel<<<NW / 256, 256, 0, stream>>>(Wv, wvt);
    cvt_t_bf16_kernel<<<NW / 256, 256, 0, stream>>>(Wo, wot);

    qkv_rope_kernel<<<dim3(256, 48), 128, 0, stream>>>(
        xb, wqt, wkt, wvt, bq, bk, bv, qbf, kbf, vtb);

    const int smem = 16 * 1024 * 4 + 16 * 1024 * 2;    // 96 KB (fits 320 KB WGP LDS)
    static_cast<void>(hipFuncSetAttribute((const void*)attn_kernel,
        hipFuncAttributeMaxDynamicSharedMemorySize, smem));
    attn_kernel<<<dim3(B_ * H_, S_ / 16), 32, smem, stream>>>(qbf, kbf, vtb, mask, att);

    oproj_kernel<<<dim3(256, 16), 128, 0, stream>>>(att, wot, bo, mask, out);
}
